// MultiHeadSelfAttention_89670327206093
// MI455X (gfx1250) — matrix operand order repaired and checked by reading, not compiled, not run on hardware
//
#include <hip/hip_runtime.h>
#include <hip/hip_bf16.h>

// Problem constants (match reference)
#define B_ 4
#define T_ 2048
#define D_ 384
#define H_ 6
#define HD_ 64
#define HALF_ 32
#define ROWS_ (B_ * T_)   // 8192

typedef __attribute__((ext_vector_type(16))) _Float16 v16h;
typedef __attribute__((ext_vector_type(8)))  _Float16 v8h;
typedef __attribute__((ext_vector_type(8)))  float    v8f;
typedef __attribute__((ext_vector_type(4)))  int      v4i;

union V16U { v16h v; v8h h[2]; };

static __device__ __forceinline__ v8f wmma_f16(v16h a, v16h b, v8f c) {
    return __builtin_amdgcn_wmma_f32_16x16x32_f16(
        /*neg_a=*/false, a, /*neg_b=*/false, b,
        /*c_mod=*/(short)0, c, /*reuse_a=*/false, /*reuse_b=*/false);
}

// ---------------------------------------------------------------------------
// gfx1250 async global->LDS staging (ASYNCcnt-tracked), with fallback
// ---------------------------------------------------------------------------
#if defined(__AMDGCN__) && __has_builtin(__builtin_amdgcn_global_load_async_to_lds_b128)
#define USE_ASYNC_LDS 1
static __device__ __forceinline__ void async_copy_b128(void* lds, const void* g) {
    __builtin_amdgcn_global_load_async_to_lds_b128(
        (__attribute__((address_space(1))) v4i*)(const_cast<void*>(g)),
        (__attribute__((address_space(3))) v4i*)lds, /*offset=*/0, /*cpol=*/0);
}
static __device__ __forceinline__ void wait_async() {
#if __has_builtin(__builtin_amdgcn_s_wait_asynccnt)
    __builtin_amdgcn_s_wait_asynccnt(0);
#else
    asm volatile("s_wait_asynccnt 0x0" ::: "memory");
#endif
}
#else
#define USE_ASYNC_LDS 0
static __device__ __forceinline__ void async_copy_b128(void* lds, const void* g) {
    *(v8h*)lds = *(const v8h*)g;
}
static __device__ __forceinline__ void wait_async() {}
#endif

// ---------------------------------------------------------------------------
// Kernel 0: convert fp32 weights to f16 in workspace
// ---------------------------------------------------------------------------
__global__ void cvt_weights_kernel(const float* __restrict__ w_qkv,
                                   const float* __restrict__ w_proj,
                                   _Float16* __restrict__ wqkv_h,
                                   _Float16* __restrict__ wproj_h) {
    int i = blockIdx.x * 256 + threadIdx.x;
    if (i < D_ * 3 * D_) wqkv_h[i] = (_Float16)w_qkv[i];
    if (i < D_ * D_)     wproj_h[i] = (_Float16)w_proj[i];
}

// ---------------------------------------------------------------------------
// Tiled WMMA GEMM: C[M,N] = A[M,K] * Bw[K,N]
// 256 threads = 8 waves. Block tile 128x64, BLK_K=32.
// Wave tile 32x32: 2 A-frags x 2 B-frags -> 4 WMMAs per K-step.
// TA: float (convert on stage) or _Float16 (async-staged). TC: f16/f32 out.
// ---------------------------------------------------------------------------
template <typename TA, typename TC>
__global__ __launch_bounds__(256) void gemm_wmma_kernel(
        const TA* __restrict__ A, const _Float16* __restrict__ Bw,
        TC* __restrict__ C, int M, int N, int K) {
    __shared__ alignas(16) _Float16 Ash[128][40];  // [m][k], padded
    __shared__ alignas(16) _Float16 Bsh[64][40];   // transposed: [n][k], padded

    const int tid  = threadIdx.x;
    const int wave = tid >> 5;
    const int lane = tid & 31;
    const int lm   = lane & 15;
    const int wr   = wave & 3;   // row group 0..3 (32 rows each)
    const int wc   = wave >> 2;  // col group 0..1 (32 cols each)
    const int rowBase = blockIdx.x * 128;
    const int colBase = blockIdx.y * 64;

    const int kbA = (lane < 16) ? 0 : 16;   // A-fragment K sub-base
    const int kbB = (lane < 16) ? 0 : 16;   // B-fragment K sub-base

    v8f acc[2][2] = {};

    for (int kk = 0; kk < K; kk += 32) {
        // Stage A tile 128x32
        if constexpr (sizeof(TA) == 2) {
            #pragma unroll
            for (int rr = 0; rr < 2; ++rr) {
                int r  = rr * 64 + (tid >> 2);
                int cc = (tid & 3) * 8;
                const TA* src = A + (size_t)(rowBase + r) * K + kk + cc;
                async_copy_b128(&Ash[r][cc], src);
            }
        } else {
            #pragma unroll
            for (int rr = 0; rr < 2; ++rr) {
                int r  = rr * 64 + (tid >> 2);
                int cc = (tid & 3) * 8;
                const TA* src = A + (size_t)(rowBase + r) * K + kk + cc;
                #pragma unroll
                for (int j = 0; j < 8; ++j) Ash[r][cc + j] = (_Float16)src[j];
            }
        }
        // Stage B tile 32x64, stored transposed [n][k]
        {
            int kr = tid >> 3;
            int cc = (tid & 7) * 8;
            const _Float16* src = Bw + (size_t)(kk + kr) * N + colBase + cc;
            #pragma unroll
            for (int j = 0; j < 8; ++j) Bsh[cc + j][kr] = src[j];
        }
        if constexpr (sizeof(TA) == 2) wait_async();
        __syncthreads();

        V16U a0, a1, b0, b1;
        a0.h[0] = *(const v8h*)&Ash[wr * 32 + lm][kbA];
        a0.h[1] = *(const v8h*)&Ash[wr * 32 + lm][kbA + 8];
        a1.h[0] = *(const v8h*)&Ash[wr * 32 + 16 + lm][kbA];
        a1.h[1] = *(const v8h*)&Ash[wr * 32 + 16 + lm][kbA + 8];
        b0.h[0] = *(const v8h*)&Bsh[wc * 32 + lm][kbB];
        b0.h[1] = *(const v8h*)&Bsh[wc * 32 + lm][kbB + 8];
        b1.h[0] = *(const v8h*)&Bsh[wc * 32 + 16 + lm][kbB];
        b1.h[1] = *(const v8h*)&Bsh[wc * 32 + 16 + lm][kbB + 8];

        acc[0][0] = wmma_f16(a0.v, b0.v, acc[0][0]);
        acc[0][1] = wmma_f16(a0.v, b1.v, acc[0][1]);
        acc[1][0] = wmma_f16(a1.v, b0.v, acc[1][0]);
        acc[1][1] = wmma_f16(a1.v, b1.v, acc[1][1]);
        __syncthreads();
    }

    // Epilogue: C layout — lane col n=lm, slot i -> row i + (lane>=16 ? 8 : 0)
    const int mb = (lane < 16) ? 0 : 8;
    #pragma unroll
    for (int mi = 0; mi < 2; ++mi) {
        #pragma unroll
        for (int i = 0; i < 8; ++i) {
            size_t r   = (size_t)(rowBase + wr * 32 + mi * 16 + mb + i);
            size_t cc0 = (size_t)(colBase + wc * 32 + lm);
            C[r * N + cc0]      = (TC)acc[mi][0][i];
            C[r * N + cc0 + 16] = (TC)acc[mi][1][i];
        }
    }
}

// ---------------------------------------------------------------------------
// Kernel 2: RoPE on Q,K + repack QKV into head-major f16 [B,H,T,HD]
// ---------------------------------------------------------------------------
__global__ __launch_bounds__(256) void rope_pack_kernel(
        const _Float16* __restrict__ qkv, const float* __restrict__ cosT,
        const float* __restrict__ sinT, _Float16* __restrict__ Qh,
        _Float16* __restrict__ Kh, _Float16* __restrict__ Vh) {
    int idx = blockIdx.x * blockDim.x + threadIdx.x;
    if (idx >= B_ * H_ * T_) return;
    int t = idx % T_;
    int h = (idx / T_) % H_;
    int b = idx / (T_ * H_);

    size_t row  = (size_t)(b * T_ + t) * (3 * D_);
    size_t qoff = row + h * HD_;
    size_t koff = row + D_ + h * HD_;
    size_t voff = row + 2 * D_ + h * HD_;
    size_t dst  = ((size_t)(b * H_ + h) * T_ + t) * HD_;

    for (int d = 0; d < HALF_; ++d) {
        float c = cosT[t * HALF_ + d];
        float s = sinT[t * HALF_ + d];
        float q1 = (float)qkv[qoff + d], q2 = (float)qkv[qoff + d + HALF_];
        Qh[dst + d]         = (_Float16)(q1 * c - q2 * s);
        Qh[dst + d + HALF_] = (_Float16)(q2 * c + q1 * s);
        float k1 = (float)qkv[koff + d], k2 = (float)qkv[koff + d + HALF_];
        Kh[dst + d]         = (_Float16)(k1 * c - k2 * s);
        Kh[dst + d + HALF_] = (_Float16)(k2 * c + k1 * s);
    }
    for (int d = 0; d < HD_; ++d) Vh[dst + d] = qkv[voff + d];
}

// ---------------------------------------------------------------------------
// Kernel 3: flash attention. Block = 8 waves; one block per (b,h,128 q-rows).
// K chunk async-staged to LDS; V staged transposed; scores 2x2 WMMAs; online
// softmax via half-wave shuffles; P re-laid out via per-wave LDS; PV 4 WMMAs.
// ---------------------------------------------------------------------------
__global__ __launch_bounds__(256) void flash_attn_kernel(
        const _Float16* __restrict__ Qh, const _Float16* __restrict__ Kh,
        const _Float16* __restrict__ Vh, const float* __restrict__ bias,
        _Float16* __restrict__ attn_h) {
    __shared__ alignas(16) _Float16 Ksh[32][64];      // [key_local][d]
    __shared__ alignas(16) _Float16 Vsht[64][32];     // transposed [d][key_local]
    __shared__ alignas(16) _Float16 Psh[8][16][32];   // per-wave P tile [m][k]

    const int tid  = threadIdx.x;
    const int wave = tid >> 5;
    const int lane = tid & 31;
    const int lm   = lane & 15;
    const int bh   = blockIdx.x >> 4;     // 0..23
    const int qg   = blockIdx.x & 15;     // 0..15
    const int b    = bh / H_;
    const int h    = bh % H_;
    const int tq0  = qg * 128 + wave * 16;

    const size_t headBase = (size_t)(b * H_ + h) * T_ * HD_;
    const int kbA = (lane < 16) ? 0 : 16;
    const int kbB = (lane < 16) ? 0 : 16;
    const int mb  = (lane < 16) ? 0 : 8;

    // Q fragments: d 0..31 and d 32..63
    V16U aq0, aq1;
    {
        const _Float16* qrow = Qh + headBase + (size_t)(tq0 + lm) * HD_;
        aq0.h[0] = *(const v8h*)(qrow + kbA);
        aq0.h[1] = *(const v8h*)(qrow + kbA + 8);
        aq1.h[0] = *(const v8h*)(qrow + 32 + kbA);
        aq1.h[1] = *(const v8h*)(qrow + 32 + kbA + 8);
    }

    v8f co[4] = {};
    float mrow[8], lrow[8];
    #pragma unroll
    for (int i = 0; i < 8; ++i) { mrow[i] = -3.0e38f; lrow[i] = 0.0f; }
    const float scale = 0.125f;   // 1/sqrt(64)

    for (int kt = 0; kt < T_ / 32; ++kt) {
        // Stage K chunk (async) and transposed V chunk
        {
            int r  = tid >> 3;
            int cc = (tid & 7) * 8;
            const _Float16* ks = Kh + headBase + (size_t)(kt * 32 + r) * HD_ + cc;
            async_copy_b128(&Ksh[r][cc], ks);
            const _Float16* vs = Vh + headBase + (size_t)(kt * 32 + r) * HD_ + cc;
            #pragma unroll
            for (int j = 0; j < 8; ++j) Vsht[cc + j][r] = vs[j];
        }
        wait_async();
        __syncthreads();

        // --- scores: S = Q K^T (16q x 32k), two 16-key WMMA tiles ---
        float s0a[8], s1a[8];
        #pragma unroll
        for (int st = 0; st < 2; ++st) {
            V16U bl, bh2;
            const _Float16* krow = &Ksh[st * 16 + lm][0];
            bl.h[0]  = *(const v8h*)(krow + kbB);
            bl.h[1]  = *(const v8h*)(krow + kbB + 8);
            bh2.h[0] = *(const v8h*)(krow + 32 + kbB);
            bh2.h[1] = *(const v8h*)(krow + 32 + kbB + 8);
            v8f s = {};
            s = wmma_f16(aq0.v, bl.v, s);
            s = wmma_f16(aq1.v, bh2.v, s);
            int kcol = kt * 32 + st * 16 + lm;
            #pragma unroll
            for (int i = 0; i < 8; ++i) {
                float sv = s[i] * scale +
                           bias[(size_t)(tq0 + mb + i) * T_ + kcol];
                if (st == 0) s0a[i] = sv; else s1a[i] = sv;
            }
        }

        // --- online softmax (row reductions across 16-lane halves) ---
        float mn[8], alpha[8];
        #pragma unroll
        for (int i = 0; i < 8; ++i) {
            float mx = fmaxf(s0a[i], s1a[i]);
            #pragma unroll
            for (int d = 1; d < 16; d <<= 1)
                mx = fmaxf(mx, __shfl_xor(mx, d, 32));
            mx = fmaxf(mx, mrow[i]);
            mn[i]    = mx;
            alpha[i] = __expf(mrow[i] - mx);
            mrow[i]  = mx;
        }
        float p0[8], p1[8];
        #pragma unroll
        for (int i = 0; i < 8; ++i) {
            p0[i] = __expf(s0a[i] - mn[i]);
            p1[i] = __expf(s1a[i] - mn[i]);
            float r = p0[i] + p1[i];
            #pragma unroll
            for (int d = 1; d < 16; d <<= 1)
                r += __shfl_xor(r, d, 32);
            lrow[i] = lrow[i] * alpha[i] + r;
            #pragma unroll
            for (int dt = 0; dt < 4; ++dt) co[dt][i] *= alpha[i];
        }

        // --- P: C layout -> LDS -> A-fragment layout ---
        #pragma unroll
        for (int i = 0; i < 8; ++i) {
            Psh[wave][mb + i][lm]      = (_Float16)p0[i];
            Psh[wave][mb + i][16 + lm] = (_Float16)p1[i];
        }
        __builtin_amdgcn_wave_barrier();
        V16U pf;
        pf.h[0] = *(const v8h*)&Psh[wave][lm][kbA];
        pf.h[1] = *(const v8h*)&Psh[wave][lm][kbA + 8];

        // --- O += P V : 4 d-tiles of 16 ---
        #pragma unroll
        for (int dt = 0; dt < 4; ++dt) {
            V16U bv;
            const _Float16* vcol = &Vsht[dt * 16 + lm][0];
            bv.h[0] = *(const v8h*)(vcol + kbB);
            bv.h[1] = *(const v8h*)(vcol + kbB + 8);
            co[dt] = wmma_f16(pf.v, bv.v, co[dt]);
        }
        __syncthreads();
    }

    // Epilogue: normalize and scatter to [B,T, h*64 + d] (f16)
    #pragma unroll
    for (int i = 0; i < 8; ++i) {
        float inv = 1.0f / lrow[i];
        int t = tq0 + mb + i;
        size_t orow = ((size_t)b * T_ + t) * D_ + h * HD_;
        #pragma unroll
        for (int dt = 0; dt < 4; ++dt)
            attn_h[orow + dt * 16 + lm] = (_Float16)(co[dt][i] * inv);
    }
}

// ---------------------------------------------------------------------------
// Launch
// ---------------------------------------------------------------------------
extern "C" void kernel_launch(void* const* d_in, const int* in_sizes, int n_in,
                              void* d_out, int out_size, void* d_ws, size_t ws_size,
                              hipStream_t stream) {
    const float* x      = (const float*)d_in[0];
    const float* bias   = (const float*)d_in[1];
    const float* cosT   = (const float*)d_in[2];
    const float* sinT   = (const float*)d_in[3];
    const float* w_qkv  = (const float*)d_in[4];
    const float* w_proj = (const float*)d_in[5];
    float* out = (float*)d_out;

    // Workspace carve-up (f16 elements)
    _Float16* wqkv_h  = (_Float16*)d_ws;                 // 384*1152
    _Float16* wproj_h = wqkv_h  + (size_t)D_ * 3 * D_;   // 384*384
    _Float16* qkv_h   = wproj_h + (size_t)D_ * D_;       // 8192*1152
    _Float16* Qh      = qkv_h   + (size_t)ROWS_ * 3 * D_;
    _Float16* Kh      = Qh      + (size_t)B_ * H_ * T_ * HD_;
    _Float16* Vh      = Kh      + (size_t)B_ * H_ * T_ * HD_;
    _Float16* attn_h  = Vh      + (size_t)B_ * H_ * T_ * HD_;

    // 0) weights -> f16
    cvt_weights_kernel<<<(D_ * 3 * D_ + 255) / 256, 256, 0, stream>>>(
        w_qkv, w_proj, wqkv_h, wproj_h);

    // 1) qkv = x @ w_qkv   (8192 x 1152, K=384)
    gemm_wmma_kernel<float, _Float16><<<dim3(ROWS_ / 128, (3 * D_) / 64), 256, 0, stream>>>(
        x, wqkv_h, qkv_h, ROWS_, 3 * D_, D_);

    // 2) RoPE + head-major repack
    rope_pack_kernel<<<(B_ * H_ * T_ + 255) / 256, 256, 0, stream>>>(
        qkv_h, cosT, sinT, Qh, Kh, Vh);

    // 3) flash attention
    flash_attn_kernel<<<B_ * H_ * (T_ / 128), 256, 0, stream>>>(
        Qh, Kh, Vh, bias, attn_h);

    // 4) out = attn @ w_proj  (8192 x 384, K=384)
    gemm_wmma_kernel<_Float16, float><<<dim3(ROWS_ / 128, D_ / 64), 256, 0, stream>>>(
        attn_h, wproj_h, out, ROWS_, D_, D_);
}
